// WordScoreEncoder_88536455840207
// MI455X (gfx1250) — matrix operand order repaired and checked by reading, not compiled, not run on hardware
//
#include <hip/hip_runtime.h>
#include <hip/hip_bf16.h>

typedef __attribute__((ext_vector_type(16))) _Float16 v16h;
typedef __attribute__((ext_vector_type(8)))  _Float16 v8h;
typedef __attribute__((ext_vector_type(4)))  _Float16 v4h;
typedef __attribute__((ext_vector_type(8)))  float    v8f;

#define WMMA_F16(a,b,c) __builtin_amdgcn_wmma_f32_16x16x32_f16(false,(a),false,(b),(short)0,(c),false,false)

// ---------------------------------------------------------------------------
// Generic implicit-GEMM conv1d (NHC layout, OIH weights), one 16x16 C tile per
// wave, K = TAPS*256 chunked by 32, f16 WMMA with f32 accumulate.
// TAPS/PAD are compile-time so the whole K loop fully unrolls (TAPS*8 WMMAs).
// grid = (Tph/16, H/16/8, B), block = 256 (8 waves)
// ---------------------------------------------------------------------------
template <int TAPS, int PAD>
__global__ __launch_bounds__(256) void conv_wmma_kernel(
    const float* __restrict__ in, const float* __restrict__ wgt,
    const float* __restrict__ bias, const float* __restrict__ resid,
    float* __restrict__ out)
{
    const int H = 256, T = 512;
    const int wave = threadIdx.x >> 5;
    const int lane = threadIdx.x & 31;
    const int lane_hi = lane >> 4;   // 0 or 1
    const int ln = lane & 15;        // N index / M row index
    const int t0 = blockIdx.x * 16;
    const int o0 = (blockIdx.y * 8 + wave) * 16;
    const int b  = blockIdx.z;

    v8f acc;
    const float bo = bias[o0 + ln];
    #pragma unroll
    for (int r = 0; r < 8; ++r) acc[r] = bo;

    #pragma unroll
    for (int k = 0; k < TAPS; ++k) {
        const int t = t0 + ln + k - PAD;   // A row's time index
        const bool inb = (t >= 0) && (t < T);
        const float* row = in + ((size_t)b * T + (inb ? t : 0)) * H;
        #pragma unroll
        for (int c0 = 0; c0 < H; c0 += 32) {
            // ---- A fragment: 16x32 f16 (contiguous K: k = 16*lane_hi + j) ----
            v16h a;
            if (inb) {
                const float4* p1 = (const float4*)(row + c0 + 16 * lane_hi);
                const float4* p2 = (const float4*)(row + c0 + 16 * lane_hi + 8);
                float4 f0 = p1[0], f1 = p1[1], f2 = p2[0], f3 = p2[1];
                a[0]=(_Float16)f0.x; a[1]=(_Float16)f0.y; a[2]=(_Float16)f0.z; a[3]=(_Float16)f0.w;
                a[4]=(_Float16)f1.x; a[5]=(_Float16)f1.y; a[6]=(_Float16)f1.z; a[7]=(_Float16)f1.w;
                a[8]=(_Float16)f2.x; a[9]=(_Float16)f2.y; a[10]=(_Float16)f2.z; a[11]=(_Float16)f2.w;
                a[12]=(_Float16)f3.x; a[13]=(_Float16)f3.y; a[14]=(_Float16)f3.z; a[15]=(_Float16)f3.w;
            } else {
                #pragma unroll
                for (int j = 0; j < 16; ++j) a[j] = (_Float16)0.0f;
            }
            // ---- B fragment: 32x16 f16 (lane = N, K consecutive) ----
            v16h bf;
            const int o = o0 + ln;
            const float* wrow = wgt + ((size_t)o * H + c0 + 16 * lane_hi) * TAPS + k;
            #pragma unroll
            for (int j = 0; j < 16; ++j) bf[j] = (_Float16)wrow[j * TAPS];

            acc = WMMA_F16(a, bf, acc);
        }
    }
    const bool has_res = (resid != nullptr);
    #pragma unroll
    for (int r = 0; r < 8; ++r) {
        const int mm = r + 8 * lane_hi;
        const size_t idx = ((size_t)b * T + (t0 + mm)) * H + o0 + ln;
        float v = acc[r];
        if (has_res) v += resid[idx];
        out[idx] = v;
    }
}

// ---------------------------------------------------------------------------
// Channel LayerNorm (+optional ReLU) over H=256, one block per (b,t)
// ---------------------------------------------------------------------------
__global__ __launch_bounds__(256) void chan_ln_kernel(
    const float* __restrict__ y, const float* __restrict__ g,
    const float* __restrict__ bv, float* __restrict__ out, int do_relu)
{
    __shared__ float s[256];
    const int bt = blockIdx.x, c = threadIdx.x;
    const float v = y[(size_t)bt * 256 + c];
    s[c] = v; __syncthreads();
    for (int off = 128; off > 0; off >>= 1) { if (c < off) s[c] += s[c + off]; __syncthreads(); }
    const float mu = s[0] * (1.0f / 256.0f);
    __syncthreads();
    const float d = v - mu;
    s[c] = d * d; __syncthreads();
    for (int off = 128; off > 0; off >>= 1) { if (c < off) s[c] += s[c + off]; __syncthreads(); }
    const float var = s[0] * (1.0f / 256.0f);
    float o = d * rsqrtf(var + 1e-4f) * g[c] + bv[c];
    if (do_relu) o = fmaxf(o, 0.0f);
    out[(size_t)bt * 256 + c] = o;
}

// ---------------------------------------------------------------------------
// gauss channel-0 dot product -> m = relu(g0)+1, one block per (b,t)
// ---------------------------------------------------------------------------
__global__ __launch_bounds__(256) void gauss_m_kernel(
    const float* __restrict__ h, const float* __restrict__ gw,
    const float* __restrict__ gb, float* __restrict__ m)
{
    __shared__ float s[256];
    const int bt = blockIdx.x, c = threadIdx.x;
    s[c] = h[(size_t)bt * 256 + c] * gw[c];  // gauss_w row 0
    __syncthreads();
    for (int off = 128; off > 0; off >>= 1) { if (c < off) s[c] += s[c + off]; __syncthreads(); }
    if (c == 0) m[bt] = fmaxf(s[0] + gb[0], 0.0f) + 1.0f;
}

__global__ void zero_kernel(float* p, int n) {
    int i = blockIdx.x * 256 + threadIdx.x;
    if (i < n) p[i] = 0.0f;
}

__global__ void scatter_dur_kernel(const int* __restrict__ x2w,
                                   const float* __restrict__ m, float* dur)
{
    int i = blockIdx.x * 256 + threadIdx.x;
    if (i < 8 * 512) { int b = i >> 9; atomicAdd(&dur[b * 257 + x2w[i]], m[i]); }
}

__global__ void scatter_durw_kernel(const int* __restrict__ m2w, float* durw)
{
    int i = blockIdx.x * 256 + threadIdx.x;
    if (i < 8 * 4096) { int b = i >> 12; atomicAdd(&durw[b * 257 + m2w[i]], 1.0f); }
}

__global__ void write_mword_kernel(const float* __restrict__ dur, float* __restrict__ out)
{
    int i = blockIdx.x * 256 + threadIdx.x;
    if (i < 8 * 256) { int b = i >> 8; out[i] = dur[b * 257 + 1 + (i & 255)]; }
}

// m_new = m * dur_word[w] / (dur[w] + 1e-4), w = x2word (fwd of stop_gradient form)
__global__ void rescale_m_kernel(const int* __restrict__ x2w, const float* __restrict__ m,
                                 const float* __restrict__ dur, const float* __restrict__ durw,
                                 float* __restrict__ mnew)
{
    int i = blockIdx.x * 256 + threadIdx.x;
    if (i < 8 * 512) {
        int b = i >> 9, w = x2w[i];
        mnew[i] = m[i] * (durw[b * 257 + w] / (dur[b * 257 + w] + 1e-4f));
    }
}

// center = inclusive_cumsum(m) - m/2, one block (512 threads) per batch
__global__ __launch_bounds__(512) void scan_center_kernel(
    const float* __restrict__ mnew, float* __restrict__ center)
{
    __shared__ float s[512];
    const int b = blockIdx.x, t = threadIdx.x;
    const float v = mnew[b * 512 + t];
    s[t] = v; __syncthreads();
    for (int off = 1; off < 512; off <<= 1) {
        float add = (t >= off) ? s[t - off] : 0.0f;
        __syncthreads();
        s[t] += add;
        __syncthreads();
    }
    center[b * 512 + t] = s[t] - 0.5f * v;
}

// ---------------------------------------------------------------------------
// Attention: per block = 16 mel rows x all 256 channels.
// Phase 1: masked logits + wave-level softmax (weights -> HBM f32, LDS f16).
// Phase 2: attn = W(16x512) @ X(512x256) via WMMA.
//   K outermost; each 32x256 X chunk is cooperatively staged transposed in
//   LDS as f16 (coalesced global_load_b32 in, packed ds_store_b64), then each
//   wave reads A (2x ds_load_b128 from softmax weights) and two B fragments
//   (contiguous K halves -> ds_load_b128 pairs) and issues 2 WMMAs.
// grid = (Tmel/16, B), block = 256
// ---------------------------------------------------------------------------
__global__ __launch_bounds__(256) void attn_wmma_kernel(
    const float* __restrict__ x, const int* __restrict__ x2w,
    const int* __restrict__ m2w, const float* __restrict__ center,
    float* __restrict__ w_out, float* __restrict__ attn_out)
{
    const int Tph = 512, Tmel = 4096, H = 256;
    __shared__ float s_c[512];
    __shared__ int   s_xw[512];
    __shared__ int   s_mw[16];
    __shared__ float s_w[16 * 512];
    __shared__ __align__(16) _Float16 s_wh[16 * 512];   // softmax weights, f16
    __shared__ __align__(16) _Float16 s_xT[256 * 32];   // X chunk, [n][k] f16

    const int tid = threadIdx.x;
    const int wave = tid >> 5, lane = tid & 31;
    const int lane_hi = lane >> 4, ln = lane & 15;
    const int b = blockIdx.y;
    const int tm0 = blockIdx.x * 16;

    for (int i = tid; i < 512; i += 256) {
        s_c[i]  = center[b * 512 + i];
        s_xw[i] = x2w[b * 512 + i];
    }
    if (tid < 16) s_mw[tid] = m2w[(size_t)b * 4096 + tm0 + tid];
    __syncthreads();

    // ---- softmax rows: wave w owns rows 2w, 2w+1 ----
    #pragma unroll
    for (int rr = 0; rr < 2; ++rr) {
        const int r  = wave * 2 + rr;
        const int tm = tm0 + r;
        const int mw = s_mw[r];
        const float tmf = (float)tm;
        float lmax = -3.0e38f;
        for (int p = lane; p < 512; p += 32) {
            float d = s_c[p] - tmf;
            float v = -d * d * 0.1f;
            if (s_xw[p] != mw) v -= 1.0e9f;
            s_w[r * 512 + p] = v;
            lmax = fmaxf(lmax, v);
        }
        #pragma unroll
        for (int off = 16; off > 0; off >>= 1) lmax = fmaxf(lmax, __shfl_xor(lmax, off, 32));
        float lsum = 0.0f;
        for (int p = lane; p < 512; p += 32) lsum += __expf(s_w[r * 512 + p] - lmax);
        #pragma unroll
        for (int off = 16; off > 0; off >>= 1) lsum += __shfl_xor(lsum, off, 32);
        const float inv = 1.0f / lsum;
        for (int p = lane; p < 512; p += 32) {
            float wv = __expf(s_w[r * 512 + p] - lmax) * inv;
            s_wh[r * 512 + p] = (_Float16)wv;
            w_out[((size_t)b * Tmel + tm) * Tph + p] = wv;
        }
    }

    // ---- WMMA phase: wave owns channels [wave*32, wave*32+32) ----
    const int n0 = wave * 32;
    v8f acc0, acc1;
    #pragma unroll
    for (int r = 0; r < 8; ++r) { acc0[r] = 0.0f; acc1[r] = 0.0f; }

    for (int k0 = 0; k0 < 512; k0 += 32) {
        __syncthreads();   // s_xT reuse / fresh chunk
        // stage X[k0:k0+32, 0:256] transposed into LDS as f16
        for (int i = tid; i < 2048; i += 256) {
            const int kq = (i >> 8) << 2;      // k offset, multiple of 4
            const int c  = i & 255;            // channel (coalesced across lanes)
            const float* xp = x + ((size_t)b * Tph + k0 + kq) * H + c;
            v4h pk;
            pk[0] = (_Float16)xp[0 * H];
            pk[1] = (_Float16)xp[1 * H];
            pk[2] = (_Float16)xp[2 * H];
            pk[3] = (_Float16)xp[3 * H];
            *(v4h*)&s_xT[c * 32 + kq] = pk;
            if (k0 + 32 < 512)
                __builtin_prefetch(xp + 32 * H, 0, 1);   // next chunk
        }
        __syncthreads();

        // A fragment from softmax weights (shared by both N-tiles)
        v8h alo = *(const v8h*)&s_wh[ln * 512 + k0 + 16 * lane_hi];
        v8h ahi = *(const v8h*)&s_wh[ln * 512 + k0 + 16 * lane_hi + 8];
        v16h a;
        #pragma unroll
        for (int j = 0; j < 8; ++j) { a[j] = alo[j]; a[8 + j] = ahi[j]; }

        // B fragments: contiguous K halves per lane
        const _Float16* p0 = &s_xT[(n0 + ln) * 32 + 16 * lane_hi];
        const _Float16* p1 = &s_xT[(n0 + 16 + ln) * 32 + 16 * lane_hi];
        v8h b0l = *(const v8h*)p0,     b0h = *(const v8h*)(p0 + 8);
        v8h b1l = *(const v8h*)p1,     b1h = *(const v8h*)(p1 + 8);
        v16h b0, b1;
        #pragma unroll
        for (int j = 0; j < 8; ++j) {
            b0[j] = b0l[j]; b0[8 + j] = b0h[j];
            b1[j] = b1l[j]; b1[8 + j] = b1h[j];
        }
        acc0 = WMMA_F16(a, b0, acc0);
        acc1 = WMMA_F16(a, b1, acc1);
    }

    #pragma unroll
    for (int r = 0; r < 8; ++r) {
        const int tm = tm0 + r + 8 * lane_hi;
        float* dst = attn_out + ((size_t)b * Tmel + tm) * H + ln;
        dst[n0]      = acc0[r];
        dst[n0 + 16] = acc1[r];
    }
}

// ---------------------------------------------------------------------------
extern "C" void kernel_launch(void* const* d_in, const int* in_sizes, int n_in,
                              void* d_out, int out_size, void* d_ws, size_t ws_size,
                              hipStream_t stream)
{
    (void)in_sizes; (void)n_in; (void)out_size; (void)ws_size;
    const float* x       = (const float*)d_in[0];
    const float* pre_w1  = (const float*)d_in[2];
    const float* pre_b1  = (const float*)d_in[3];
    const float* ln1_g   = (const float*)d_in[4];
    const float* ln1_b   = (const float*)d_in[5];
    const float* pre_w2  = (const float*)d_in[6];
    const float* pre_b2  = (const float*)d_in[7];
    const float* ln2_g   = (const float*)d_in[8];
    const float* ln2_b   = (const float*)d_in[9];
    const float* proj_w  = (const float*)d_in[10];
    const float* proj_b  = (const float*)d_in[11];
    const float* gauss_w = (const float*)d_in[12];
    const float* gauss_b = (const float*)d_in[13];
    const int*   x2word  = (const int*)d_in[14];
    const int*   mel2wd  = (const int*)d_in[15];

    const size_t ACT = (size_t)8 * 512 * 256;   // 1,048,576 floats
    float* ws   = (float*)d_ws;
    float* bufA = ws;
    float* bufB = bufA + ACT;
    float* m    = bufB + ACT;          // 4096
    float* mnew = m + 4096;            // 4096
    float* cen  = mnew + 4096;         // 4096
    float* dur  = cen + 4096;          // 2560 (8*257 used)
    float* durw = dur + 2560;          // 2560

    float* out      = (float*)d_out;
    float* attn_out = out;                                  // 8*4096*256
    float* w_out    = out + (size_t)8 * 4096 * 256;         // 8*4096*512
    float* mw_out   = out + (size_t)8 * 4096 * 256 + (size_t)8 * 4096 * 512;

    const dim3 cgrid(32, 2, 8);   // t-tiles x (o-tiles/8) x B

    // pre-net stage 1 / 2, proj + residual  (x_mask == 1, omitted)
    conv_wmma_kernel<5, 2><<<cgrid, 256, 0, stream>>>(x,    pre_w1, pre_b1, nullptr, bufA);
    chan_ln_kernel<<<4096, 256, 0, stream>>>(bufA, ln1_g, ln1_b, bufB, 1);
    conv_wmma_kernel<5, 2><<<cgrid, 256, 0, stream>>>(bufB, pre_w2, pre_b2, nullptr, bufA);
    chan_ln_kernel<<<4096, 256, 0, stream>>>(bufA, ln2_g, ln2_b, bufB, 1);
    conv_wmma_kernel<1, 0><<<cgrid, 256, 0, stream>>>(bufB, proj_w, proj_b, x, bufA);

    // durations
    gauss_m_kernel<<<4096, 256, 0, stream>>>(bufA, gauss_w, gauss_b, m);
    zero_kernel<<<(2560 * 2 + 255) / 256, 256, 0, stream>>>(dur, 2560 * 2);
    scatter_dur_kernel<<<16, 256, 0, stream>>>(x2word, m, dur);
    scatter_durw_kernel<<<128, 256, 0, stream>>>(mel2wd, durw);
    write_mword_kernel<<<8, 256, 0, stream>>>(dur, mw_out);
    rescale_m_kernel<<<16, 256, 0, stream>>>(x2word, m, dur, durw, mnew);
    scan_center_kernel<<<8, 512, 0, stream>>>(mnew, cen);

    // masked softmax + attn GEMM
    attn_wmma_kernel<<<dim3(256, 8), 256, 0, stream>>>(x, x2word, mel2wd, cen, w_out, attn_out);
}
